// TGAT_79980880986756
// MI455X (gfx1250) — matrix operand order repaired and checked by reading, not compiled, not run on hardware
//
#include <hip/hip_runtime.h>

// ---------------------------------------------------------------------------
// TGAT (2-layer temporal graph attention) for gfx1250.
//
// Math restructure (exact up to FP reassociation):
//   q_time_feat = cos(time_b)  (constant)  ->  folded into Q bias
//   score[h,k]  = kv_in[k] . (Wk_h @ q_h)  ->  batched GEMM "qk" + dot pass
//   out_h       = (attn_h @ kv_in) @ Wv_h  ->  bandwidth pass "U" + GEMM
// so all heavy matmuls become row-batched GEMMs done with
// v_wmma_f32_16x16x32_f16 (f16 inputs, f32 accumulate), and the per-neighbor
// work is a single fused LDS kernel (gather + Time2Vec + scores + softmax + U).
//
// GEMM is register-blocked: one A fragment is reused against NT B fragments
// (NT = 4 or 3), emitting NT v_wmma per A load and cutting A-side global
// traffic by NT x.
// ---------------------------------------------------------------------------

typedef __attribute__((ext_vector_type(16))) _Float16 v16h;
typedef __attribute__((ext_vector_type(8)))  _Float16 v8h;
typedef __attribute__((ext_vector_type(8)))  float    v8f;

// ---- problem constants ----
constexpr int BB    = 600;     // batch
constexpr int KNB   = 20;      // neighbors per node
constexpr int ND    = 128;     // node feat dim
constexpr int TD    = 100;     // time dim
constexpr int DQ    = 228;     // ND + TD
constexpr int DK    = 356;     // ND + ED + TD
constexpr int DH    = 114;     // DQ / 2 heads
constexpr int SIDE0 = 12600;   // rows per side in layer-0 batch (600 + 12000)
constexpr int N0    = 25200;   // 2 * SIDE0   (multiple of 16)
constexpr int N1    = 1200;    // 2 * 600     (multiple of 16)
constexpr int DKP   = 384;     // DK padded (mult of 32); also CAT row width
constexpr int QHP   = 256;     // per-head padded Q width (2 * 128)
constexpr int DQP   = 240;     // DQ padded to mult of 16
constexpr int QKLD  = 768;     // 2 * DKP
constexpr float SCALE_DH = 0.09365858115816944f; // 1/sqrt(114)

static __device__ __forceinline__ v8f wmma16(v16h a, v16h b, v8f c) {
  // D = A(16x32 f16) x B(32x16 f16) + C(16x16 f32)
  return __builtin_amdgcn_wmma_f32_16x16x32_f16(
      /*neg_a=*/false, a, /*neg_b=*/false, b,
      /*c_mod=*/(short)0, c, /*reuse_a=*/false, /*reuse_b=*/false);
}

// ---------------------------------------------------------------------------
// Register-blocked WMMA GEMM:
//   out[N, outPad] = A[N, kdim](f16) @ Bt[outPad, kdim](f16)^T + bias
// One wave per 16 x (16*NT) output tile; grid = (N/16, outPad/(16*NT)).
// A rows can be remapped: phys = (r / rowsPerSide)*sideStride + r%rowsPerSide.
// Fragment layouts as used here (A and B share a contiguous k order):
//   A 16x32: lane l holds row (l&15), k = (l>>4)*16 + [0..15]
//            -> two contiguous 16B loads.
//   B 32x16: lane l holds col (l&15), halves k = (l>>4)*16 + [0..15]
//            -> from packed Bt[col][k]: 32 contiguous bytes.
//   C/D: lane l, elem i -> (row = i + 8*(l>>4), col = l&15).
// ---------------------------------------------------------------------------
template <int NT>
__global__ __launch_bounds__(32)
void gemm_wmma(const _Float16* __restrict__ A, int lda, int rowsPerSide, int sideStride,
               const _Float16* __restrict__ Bt, int kdim,
               const float* __restrict__ bias, int relu,
               float* __restrict__ outF, int ldf,
               _Float16* __restrict__ outH, int ldh)
{
  const int l  = threadIdx.x;
  const int mt = blockIdx.x;
  const int nt = blockIdx.y;
  const int hi = l >> 4;      // 0 or 1 (half-wave)
  const int ln = l & 15;

  int mrow = mt * 16 + ln;
  int arow = (rowsPerSide > 0)
               ? (mrow / rowsPerSide) * sideStride + (mrow % rowsPerSide)
               : mrow;
  const _Float16* ap = A + (size_t)arow * lda + hi * 16;

  const _Float16* bp[NT];
#pragma unroll
  for (int tt = 0; tt < NT; ++tt) {
    int ncol = (nt * NT + tt) * 16 + ln;
    bp[tt] = Bt + (size_t)ncol * kdim + hi * 16;
  }

  v8f acc[NT] = {};

  for (int kk = 0; kk < kdim; kk += 32) {
    __builtin_prefetch(ap + kk + 64, 0, 1);      // global_prefetch_b8
    v8h a0 = *(const v8h*)(ap + kk);
    v8h a1 = *(const v8h*)(ap + kk + 8);
    v16h av = __builtin_shufflevector(a0, a1,
        0,1,2,3,4,5,6,7,8,9,10,11,12,13,14,15);
#pragma unroll
    for (int tt = 0; tt < NT; ++tt) {
      v8h b0 = *(const v8h*)(bp[tt] + kk);
      v8h b1 = *(const v8h*)(bp[tt] + kk + 8);
      v16h bv = __builtin_shufflevector(b0, b1,
          0,1,2,3,4,5,6,7,8,9,10,11,12,13,14,15);
      acc[tt] = wmma16(av, bv, acc[tt]);
    }
  }

#pragma unroll
  for (int tt = 0; tt < NT; ++tt) {
    int ncol = (nt * NT + tt) * 16 + ln;
    float bv = bias ? bias[ncol] : 0.f;
#pragma unroll
    for (int i = 0; i < 8; ++i) {
      int row = mt * 16 + i + 8 * hi;
      float v = acc[tt][i] + bv;
      if (relu) v = fmaxf(v, 0.f);
      if (outF) outF[(size_t)row * ldf + ncol] = v;
      if (outH) outH[(size_t)row * ldh + ncol] = (_Float16)v;
    }
  }
}

// ---------------------------------------------------------------------------
// Row -> node id for the layer-0 batch.
// Row r: side s = r/SIDE0, j = r%SIDE0.
//   j <  600 : root node     -> src/dst_ids[j]
//   j >= 600 : hop-1 node m  -> nbr_nids0.reshape(3,600,20)[s] flat at m
// ---------------------------------------------------------------------------
static __device__ __forceinline__
int l0_node_id(int r, const int* src_ids, const int* dst_ids, const int* nn0) {
  int s = r / SIDE0, j = r - s * SIDE0;
  if (j < BB) return (s ? dst_ids : src_ids)[j];
  return nn0[s * 12000 + (j - BB)];
}

// Fill CAT[:, 240:384] with f16(static feats of the row's node) and zero pad.
__global__ void gather_cat(_Float16* __restrict__ cat, int N, int mode,
                           const float* __restrict__ staticF,
                           const int* __restrict__ src_ids,
                           const int* __restrict__ dst_ids,
                           const int* __restrict__ nn0)
{
  int g = blockIdx.x * blockDim.x + threadIdx.x;
  if (g >= N * 144) return;
  int r = g / 144, c = 240 + (g - (g / 144) * 144);
  float v = 0.f;
  if (c < 368) {
    int id;
    if (mode == 0) id = l0_node_id(r, src_ids, dst_ids, nn0);
    else { int s = r / BB, j = r - s * BB; id = (s ? dst_ids : src_ids)[j]; }
    v = staticF[(size_t)id * ND + (c - 240)];
  }
  cat[(size_t)r * DKP + c] = (_Float16)v;
}

// ---------------------------------------------------------------------------
// Fused attention core: one 64-thread workgroup per row.
//  1. build kv_in[20][356] in LDS (nbr feat | edge feat | Time2Vec cos)
//  2. scores[h][k] = kv_in[k] . qk[h]  (masked, scaled)
//  3. softmax over k
//  4. U[h][c] = sum_k attn[h][k] * kv_in[k][c]   (written over qk buffer)
// mode 0: layer-0 batch (neighbor feats gathered from static table)
// mode 1: layer-1 batch (neighbor feats = layer-0 embeddings h1f32)
// ---------------------------------------------------------------------------
constexpr int KVS = 361;  // odd stride -> no LDS bank conflicts over k

__global__ __launch_bounds__(64)
void attn_core(int mode,
               const float* __restrict__ staticF, const float* __restrict__ h1f32,
               const int* __restrict__ src_ids, const int* __restrict__ dst_ids,
               const float* __restrict__ itimes,
               const int* __restrict__ nn0, const float* __restrict__ nt0,
               const float* __restrict__ nf0,
               const int* __restrict__ nn1, const float* __restrict__ nt1,
               const float* __restrict__ nf1,
               const float* __restrict__ time_w, const float* __restrict__ time_b,
               const _Float16* qk, _Float16* u)   // may alias
{
  __shared__ float kv[KNB][KVS];
  __shared__ float qks[2][DK];
  __shared__ float sc[2][KNB];
  __shared__ int   ids[KNB];
  __shared__ float tks[KNB];
  __shared__ float tw[TD], tb[TD];

  const int r = blockIdx.x;
  const int t = threadIdx.x;

  int s, j;
  long nbBase;
  const int* nids; const float* ntimes; const float* efeat;
  float tn;
  int byTable;
  if (mode == 0) {
    s = r / SIDE0; j = r - s * SIDE0;
    if (j < BB) {
      nbBase = (long)(s * BB + j) * KNB;
      nids = nn0; ntimes = nt0; efeat = nf0;
      tn = itimes[j];
    } else {
      int m = j - BB;
      nbBase = (long)(s * 12000 + m) * KNB;
      nids = nn1; ntimes = nt1; efeat = nf1;
      tn = nt0[s * 12000 + m];
    }
    byTable = 1;
  } else {
    s = r / BB; j = r - s * BB;
    nbBase = (long)(s * BB + j) * KNB;
    nids = nn0; ntimes = nt0; efeat = nf0;
    tn = itimes[j];
    byTable = 0;
  }

  if (t < KNB) { ids[t] = nids[nbBase + t]; tks[t] = ntimes[nbBase + t]; }
  for (int c = t; c < TD; c += 64) { tw[c] = time_w[c]; tb[c] = time_b[c]; }
  __syncthreads();

  // phase 1: kv_in + qk into LDS
  for (int idx = t; idx < KNB * ND; idx += 64) {
    int k = idx / ND, c = idx - k * ND;
    float nv;
    if (byTable) nv = staticF[(size_t)ids[k] * ND + c];
    else         nv = h1f32[(size_t)(s * SIDE0 + BB + j * KNB + k) * ND + c];
    kv[k][c]      = nv;
    kv[k][ND + c] = efeat[(size_t)(nbBase + k) * ND + c];
  }
  for (int idx = t; idx < KNB * TD; idx += 64) {
    int k = idx / TD, c = idx - k * TD;
    kv[k][2 * ND + c] = __cosf((tn - tks[k]) * tw[c] + tb[c]);
  }
  for (int idx = t; idx < 2 * DK; idx += 64) {
    int h = idx / DK, c = idx - h * DK;
    qks[h][c] = (float)qk[(size_t)r * QKLD + h * DKP + c];
  }
  __syncthreads();

  // phase 2: masked, scaled scores
  if (t < 2 * KNB) {
    int h = t / KNB, k = t - h * KNB;
    float acc = 0.f;
    for (int c = 0; c < DK; ++c) acc += kv[k][c] * qks[h][c];
    acc *= SCALE_DH;
    if (!(ids[k] > 0)) acc = -1e9f;
    sc[h][k] = acc;
  }
  __syncthreads();

  // phase 3: softmax over k (K=20, serial per head)
  if (t < 2) {
    float mx = -1e30f;
    for (int k = 0; k < KNB; ++k) mx = fmaxf(mx, sc[t][k]);
    float e[KNB], sum = 0.f;
    for (int k = 0; k < KNB; ++k) { e[k] = __expf(sc[t][k] - mx); sum += e[k]; }
    float inv = 1.f / sum;
    for (int k = 0; k < KNB; ++k) sc[t][k] = e[k] * inv;
  }
  __syncthreads();

  // phase 4: U = attn @ kv_in (zero-pad to DKP)
  for (int idx = t; idx < 2 * DKP; idx += 64) {
    int h = idx / DKP, c = idx - h * DKP;
    float acc = 0.f;
    if (c < DK)
      for (int k = 0; k < KNB; ++k) acc += sc[h][k] * kv[k][c];
    u[(size_t)r * QKLD + h * DKP + c] = (_Float16)acc;
  }
}

// ---------------------------------------------------------------------------
// Weight repack kernels (f32 -> f16, transposed to Bt[out][k], zero padded).
// Output flat index == global thread index by construction.
// ---------------------------------------------------------------------------
__global__ void pack_wq_k(_Float16* out, const float* wq) {  // [2][256][128]
  int g = blockIdx.x * blockDim.x + threadIdx.x;
  if (g >= 2 * QHP * 128) return;
  int l = g / (QHP * 128), rem = g - l * (QHP * 128);
  int o = rem / 128, c = rem - o * 128;
  int h = o >> 7, d = o & 127;
  float v = (d < DH) ? wq[((size_t)l * DQ + c) * DQ + h * DH + d] : 0.f;
  out[g] = (_Float16)v;
}
__global__ void pack_qbias_k(float* out, const float* wq, const float* time_b) { // [2][256]
  int g = blockIdx.x * blockDim.x + threadIdx.x;
  if (g >= 2 * QHP) return;
  int l = g / QHP, o = g - l * QHP;
  int h = o >> 7, d = o & 127;
  float acc = 0.f;
  if (d < DH)
    for (int t = 0; t < TD; ++t)
      acc += cosf(time_b[t]) * wq[((size_t)l * DQ + ND + t) * DQ + h * DH + d];
  out[g] = acc;
}
__global__ void pack_wk_k(_Float16* out, const float* wk) {  // [2][2][384][128]
  int g = blockIdx.x * blockDim.x + threadIdx.x;
  if (g >= 2 * 2 * DKP * 128) return;
  int lh = g / (DKP * 128), rem = g - lh * (DKP * 128);
  int c = rem / 128, d = rem - c * 128;
  int l = lh >> 1, h = lh & 1;
  float v = (c < DK && d < DH) ? wk[((size_t)l * DK + c) * DQ + h * DH + d] : 0.f;
  out[g] = (_Float16)v;
}
__global__ void pack_wv_k(_Float16* out, const float* wv) {  // [2][2][128][384]
  int g = blockIdx.x * blockDim.x + threadIdx.x;
  if (g >= 2 * 2 * 128 * DKP) return;
  int lh = g / (128 * DKP), rem = g - lh * (128 * DKP);
  int d = rem / DKP, c = rem - d * DKP;
  int l = lh >> 1, h = lh & 1;
  float v = (c < DK && d < DH) ? wv[((size_t)l * DK + c) * DQ + h * DH + d] : 0.f;
  out[g] = (_Float16)v;
}
__global__ void pack_wo_k(_Float16* out, const float* wo) {  // [2][240][256]
  int g = blockIdx.x * blockDim.x + threadIdx.x;
  if (g >= 2 * DQP * QHP) return;
  int l = g / (DQP * QHP), rem = g - l * (DQP * QHP);
  int jj = rem / QHP, o = rem - jj * QHP;
  int h = o >> 7, d = o & 127;
  float v = (jj < DQ && d < DH) ? wo[((size_t)l * DQ + h * DH + d) * DQ + jj] : 0.f;
  out[g] = (_Float16)v;
}
__global__ void pack_bo_k(float* out, const float* bo) {     // [2][240]
  int g = blockIdx.x * blockDim.x + threadIdx.x;
  if (g >= 2 * DQP) return;
  int l = g / DQP, jj = g - l * DQP;
  out[g] = (jj < DQ) ? bo[l * DQ + jj] : 0.f;
}
__global__ void pack_w1_k(_Float16* out, const float* w1) {  // [2][128][384]
  int g = blockIdx.x * blockDim.x + threadIdx.x;
  if (g >= 2 * 128 * DKP) return;
  int l = g / (128 * DKP), rem = g - l * (128 * DKP);
  int e = rem / DKP, k = rem - e * DKP;
  float v = 0.f;                                 // w1 is [2][356][128]
  if (k < DQ)                   v = w1[((size_t)l * 356 + k) * 128 + e];
  else if (k >= 240 && k < 368) v = w1[((size_t)l * 356 + DQ + (k - 240)) * 128 + e];
  out[g] = (_Float16)v;
}
__global__ void pack_w2_k(_Float16* out, const float* w2) {  // [2][128][128]
  int g = blockIdx.x * blockDim.x + threadIdx.x;
  if (g >= 2 * 128 * 128) return;
  int l = g / (128 * 128), rem = g - l * (128 * 128);
  int e2 = rem / 128, e = rem - e2 * 128;
  out[g] = (_Float16)w2[((size_t)l * 128 + e) * 128 + e2];
}

// ---------------------------------------------------------------------------
static void launch_gemm(hipStream_t st, const _Float16* A, int lda, int rps, int sstr,
                        const _Float16* Bt, int kdim, const float* bias, int relu,
                        float* outF, int ldf, _Float16* outH, int ldh,
                        int N, int outPad)
{
  if (outPad % 64 == 0) {
    dim3 g(N / 16, outPad / 64);
    gemm_wmma<4><<<g, 32, 0, st>>>(A, lda, rps, sstr, Bt, kdim, bias, relu,
                                   outF, ldf, outH, ldh);
  } else {            // outPad % 48 == 0 (the 240-wide case)
    dim3 g(N / 16, outPad / 48);
    gemm_wmma<3><<<g, 32, 0, st>>>(A, lda, rps, sstr, Bt, kdim, bias, relu,
                                   outF, ldf, outH, ldh);
  }
}

extern "C" void kernel_launch(void* const* d_in, const int* in_sizes, int n_in,
                              void* d_out, int out_size, void* d_ws, size_t ws_size,
                              hipStream_t stream) {
  (void)in_sizes; (void)n_in; (void)out_size; (void)ws_size;

  const int*   src_ids = (const int*)  d_in[0];
  const int*   dst_ids = (const int*)  d_in[1];
  const float* itimes  = (const float*)d_in[2];
  const float* staticF = (const float*)d_in[3];
  const int*   nn0     = (const int*)  d_in[4];
  const float* nt0     = (const float*)d_in[5];
  const float* nf0     = (const float*)d_in[6];
  const int*   nn1     = (const int*)  d_in[7];
  const float* nt1     = (const float*)d_in[8];
  const float* nf1     = (const float*)d_in[9];
  const float* time_w  = (const float*)d_in[10];
  const float* time_b  = (const float*)d_in[11];
  const float* wq      = (const float*)d_in[12];
  const float* wk      = (const float*)d_in[13];
  const float* wv      = (const float*)d_in[14];
  const float* wo      = (const float*)d_in[15];
  const float* bo      = (const float*)d_in[16];
  const float* w1      = (const float*)d_in[17];
  const float* b1      = (const float*)d_in[18];
  const float* w2      = (const float*)d_in[19];
  const float* b2      = (const float*)d_in[20];
  float* out = (float*)d_out;   // [2][600][128]

  // ---- workspace carve (~102 MB total) ----
  char* w = (char*)d_ws;
  auto carve = [&](size_t bytes) { char* p = w; w += (bytes + 255) & ~(size_t)255; return p; };
  _Float16* WqTP  = (_Float16*)carve(2 * QHP * 128 * 2);       // [l][256][128]
  _Float16* WkTP  = (_Float16*)carve(2 * 2 * DKP * 128 * 2);   // [l][h][384][128]
  _Float16* WvTP  = (_Float16*)carve(2 * 2 * 128 * DKP * 2);   // [l][h][128][384]
  _Float16* WoTP  = (_Float16*)carve(2 * DQP * QHP * 2);       // [l][240][256]
  _Float16* W1TP  = (_Float16*)carve(2 * 128 * DKP * 2);       // [l][128][384]
  _Float16* W2T   = (_Float16*)carve(2 * 128 * 128 * 2);       // [l][128][128]
  float*    qbias = (float*)   carve(2 * QHP * 4);
  float*    boP   = (float*)   carve(2 * DQP * 4);
  _Float16* CAT0  = (_Float16*)carve((size_t)N0 * DKP * 2);    // [att240 | raw128 | pad16]
  _Float16* Qb0   = (_Float16*)carve((size_t)N0 * QHP * 2);    // Q; later reused as att_pre
  _Float16* QK0   = (_Float16*)carve((size_t)N0 * QKLD * 2);   // qk; later reused as U
  _Float16* H0    = (_Float16*)carve((size_t)N0 * 128 * 2);
  float*    h1f32 = (float*)   carve((size_t)N0 * 128 * 4);
  _Float16* h1f16 = (_Float16*)carve((size_t)N0 * 128 * 2);
  _Float16* CAT1  = (_Float16*)carve((size_t)N1 * DKP * 2);
  _Float16* Qb1   = (_Float16*)carve((size_t)N1 * QHP * 2);
  _Float16* QK1   = (_Float16*)carve((size_t)N1 * QKLD * 2);
  _Float16* H1    = (_Float16*)carve((size_t)N1 * 128 * 2);

  auto g1d = [](int total) { return dim3((total + 255) / 256); };

  // ---- repack weights (f16, transposed, zero padded) ----
  pack_wq_k   <<<g1d(2 * QHP * 128),     256, 0, stream>>>(WqTP, wq);
  pack_qbias_k<<<g1d(2 * QHP),           256, 0, stream>>>(qbias, wq, time_b);
  pack_wk_k   <<<g1d(2 * 2 * DKP * 128), 256, 0, stream>>>(WkTP, wk);
  pack_wv_k   <<<g1d(2 * 2 * 128 * DKP), 256, 0, stream>>>(WvTP, wv);
  pack_wo_k   <<<g1d(2 * DQP * QHP),     256, 0, stream>>>(WoTP, wo);
  pack_bo_k   <<<g1d(2 * DQP),           256, 0, stream>>>(boP, bo);
  pack_w1_k   <<<g1d(2 * 128 * DKP),     256, 0, stream>>>(W1TP, w1);
  pack_w2_k   <<<g1d(2 * 128 * 128),     256, 0, stream>>>(W2T, w2);

  // ---- gather raw node features into CAT buffers ----
  gather_cat<<<g1d(N0 * 144), 256, 0, stream>>>(CAT0, N0, 0, staticF, src_ids, dst_ids, nn0);
  gather_cat<<<g1d(N1 * 144), 256, 0, stream>>>(CAT1, N1, 1, staticF, src_ids, dst_ids, nn0);

  // ================= layer 0 (l = 0) over N0 rows =================
  // Q = raw @ Wq_top + qbias      [N0,128] @ [128,256]
  launch_gemm(stream, CAT0 + 240, DKP, 0, 0, WqTP, 128, qbias, 0,
              nullptr, 0, Qb0, QHP, N0, QHP);
  // qk_h = Q_h @ Wk_h^T           [N0,128] @ [128,384]
  for (int h = 0; h < 2; ++h)
    launch_gemm(stream, Qb0 + h * 128, QHP, 0, 0, WkTP + (size_t)(0 * 2 + h) * DKP * 128,
                128, nullptr, 0, nullptr, 0, QK0 + h * DKP, QKLD, N0, DKP);
  // fused kv_in build + scores + softmax + U (U overwrites qk in place)
  attn_core<<<N0, 64, 0, stream>>>(0, staticF, h1f32, src_ids, dst_ids, itimes,
                                   nn0, nt0, nf0, nn1, nt1, nf1, time_w, time_b, QK0, QK0);
  // att_pre_h = U_h @ Wv_h        [N0,384] @ [384,128]   (reuse Qb0)
  for (int h = 0; h < 2; ++h)
    launch_gemm(stream, QK0 + h * DKP, QKLD, 0, 0, WvTP + (size_t)(0 * 2 + h) * 128 * DKP,
                DKP, nullptr, 0, nullptr, 0, Qb0 + h * 128, QHP, N0, 128);
  // att = att_pre @ Wo + bo       [N0,256] @ [256,240] -> CAT cols 0..239
  launch_gemm(stream, Qb0, QHP, 0, 0, WoTP, QHP, boP, 0,
              nullptr, 0, CAT0, DKP, N0, DQP);
  // merge1: relu(CAT @ W1 + b1)   [N0,384] @ [384,128]
  launch_gemm(stream, CAT0, DKP, 0, 0, W1TP, DKP, b1, 1,
              nullptr, 0, H0, 128, N0, 128);
  // merge2: H @ W2 + b2 -> h1 (f32 for gathers, f16 for layer-1 GEMM)
  launch_gemm(stream, H0, 128, 0, 0, W2T, 128, b2, 0,
              h1f32, 128, h1f16, 128, N0, 128);

  // ================= layer 1 (l = 1) over N1 rows =================
  // Q from root embeddings: A rows mapped r -> (r/600)*12600 + r%600
  launch_gemm(stream, h1f16, 128, BB, SIDE0, WqTP + (size_t)QHP * 128, 128,
              qbias + QHP, 0, nullptr, 0, Qb1, QHP, N1, QHP);
  for (int h = 0; h < 2; ++h)
    launch_gemm(stream, Qb1 + h * 128, QHP, 0, 0, WkTP + (size_t)(1 * 2 + h) * DKP * 128,
                128, nullptr, 0, nullptr, 0, QK1 + h * DKP, QKLD, N1, DKP);
  attn_core<<<N1, 64, 0, stream>>>(1, staticF, h1f32, src_ids, dst_ids, itimes,
                                   nn0, nt0, nf0, nn1, nt1, nf1, time_w, time_b, QK1, QK1);
  for (int h = 0; h < 2; ++h)
    launch_gemm(stream, QK1 + h * DKP, QKLD, 0, 0, WvTP + (size_t)(1 * 2 + h) * 128 * DKP,
                DKP, nullptr, 0, nullptr, 0, Qb1 + h * 128, QHP, N1, 128);
  launch_gemm(stream, Qb1, QHP, 0, 0, WoTP + (size_t)DQP * QHP, QHP, boP + DQP, 0,
              nullptr, 0, CAT1, DKP, N1, DQP);
  launch_gemm(stream, CAT1, DKP, 0, 0, W1TP + (size_t)128 * DKP, DKP, b1 + 128, 1,
              nullptr, 0, H1, 128, N1, 128);
  // final merge2 writes straight into d_out [2,600,128]
  launch_gemm(stream, H1, 128, 0, 0, W2T + (size_t)128 * 128, 128, b2 + 128, 0,
              out, 128, nullptr, 0, N1, 128);
}
